// NasGNN_77670188580955
// MI455X (gfx1250) — compile-verified, matrix operand order checked by reading, not run on hardware
//
#include <hip/hip_runtime.h>

typedef __attribute__((ext_vector_type(16))) __bf16 v16bf;
typedef __attribute__((ext_vector_type(8)))  float  v8f;

#define DIM   64
#define EHID  128
#define NN    8192
#define NE    65536
#define NG    64
#define KCH   32
#define NCH   4

__device__ __forceinline__ unsigned short f2bf(float x) {
  unsigned int u = __float_as_uint(x);
  u += 0x7FFFu + ((u >> 16) & 1u);          // round-to-nearest-even
  return (unsigned short)(u >> 16);
}
__device__ __forceinline__ float bf2f(unsigned short h) {
  return __uint_as_float(((unsigned int)h) << 16);
}

// ---------------- h = relu(x * w0 + b0), also bf16 copy ----------------
__global__ void __launch_bounds__(256)
k_lin0(const float* __restrict__ x, const float* __restrict__ w0,
       const float* __restrict__ b0, float* __restrict__ H,
       unsigned short* __restrict__ HB) {
  int t = blockIdx.x * 256 + threadIdx.x;          // NN*DIM
  int n = t >> 6, f = t & 63;
  float v = fmaxf(x[n] * w0[f] + b0[f], 0.0f);
  H[t] = v; HB[t] = f2bf(v);
}

// ---------------- ee = relu(edge_attr @ w1 + b1) -> bf16 ----------------
__global__ void __launch_bounds__(256)
k_edge(const float* __restrict__ EA, const float* __restrict__ W1,
       const float* __restrict__ B1, unsigned short* __restrict__ EE) {
  int t = blockIdx.x * 256 + threadIdx.x;          // NE*EHID
  int e = t >> 7, k = t & 127;
  float s = B1[k];
#pragma unroll
  for (int j = 0; j < 4; ++j) s += EA[e * 4 + j] * W1[j * 128 + k];
  EE[t] = f2bf(fmaxf(s, 0.0f));
}

// ------- pack w2[k, d*64+f] -> w2p[d, k*64+f] (bf16), i.e. GEMM B -------
__global__ void __launch_bounds__(256)
k_pack_w2(const float* __restrict__ W2, unsigned short* __restrict__ W2P) {
  int t = blockIdx.x * 256 + threadIdx.x;          // 128*4096
  int k = t >> 12, r = t & 4095, d = r >> 6, f = r & 63;
  W2P[(size_t)d * 8192 + k * 64 + f] = f2bf(W2[t]);
}

__global__ void __launch_bounds__(256)
k_tobf16(const float* __restrict__ src, unsigned short* __restrict__ dst, int n) {
  int t = blockIdx.x * 256 + threadIdx.x;
  if (t < n) dst[t] = f2bf(src[t]);
}

__global__ void __launch_bounds__(256)
k_zero(float* __restrict__ p, int n) {
  int t = blockIdx.x * blockDim.x + threadIdx.x;
  if (t < n) p[t] = 0.0f;
}

// ---------------- C[M,N] = A[M,64]bf16 @ B[64,N]bf16 via WMMA ----------------
// 256 threads = 8 waves arranged 2(M) x 4(N); each wave computes 16 x (16*NT)
// reusing its A fragment NT times per K-step. Block tile: 32 x (64*NT).
// grid = (N/(64*NT), M/32). OB: bf16 (true) or fp32 (false) output.
template <int NT, bool OB>
__global__ void __launch_bounds__(256)
k_gemm64(const unsigned short* __restrict__ A,
         const unsigned short* __restrict__ B, int bcol, int ldb,
         void* __restrict__ C, int ldc) {
  __shared__ unsigned int   As[32 * 33];               // padded
  __shared__ unsigned short Bs[64 * NT * 66];          // transposed (n-major), padded
  const int tid = threadIdx.x;
  const int m0 = blockIdx.y * 32;
  const int n0 = blockIdx.x * (64 * NT);
  const unsigned int* Ag = (const unsigned int*)A;
  const unsigned int* Bg = (const unsigned int*)B;
  const int ldbu = ldb >> 1;
  const int bcu  = (bcol + n0) >> 1;
#pragma unroll
  for (int t = 0; t < 4; ++t) {                        // A tile 32x64 bf16
    int i = tid + t * 256;
    int m = i >> 5, c = i & 31;
    As[m * 33 + c] = Ag[(size_t)(m0 + m) * 32 + c];
  }
#pragma unroll
  for (int t = 0; t < 8 * NT; ++t) {                   // B tile 64x(64*NT), transpose
    int i = tid + t * 256;
    int k = i / (32 * NT), np = i % (32 * NT);
    unsigned int v = Bg[(size_t)k * ldbu + bcu + np];
    Bs[(2 * np + 0) * 66 + k] = (unsigned short)(v & 0xFFFFu);
    Bs[(2 * np + 1) * 66 + k] = (unsigned short)(v >> 16);
  }
  __syncthreads();

  const int w = tid >> 5, l = tid & 31;
  const int mT = (w >> 2) * 16;
  const int nW = (w & 3) * (16 * NT);
  const int mr = mT + (l & 15);
  const int kh = l >> 4;

  union { v16bf v; unsigned int u[8]; } a, b;
  v8f acc[NT];
#pragma unroll
  for (int q = 0; q < NT; ++q) acc[q] = {};
#pragma unroll
  for (int kb = 0; kb < 64; kb += 32) {
#pragma unroll
    for (int j = 0; j < 8; ++j) {
      int kp = kb + ((j < 4) ? (2 * j + 8 * kh) : (16 + 2 * (j - 4) + 8 * kh));
      a.u[j] = As[mr * 33 + (kp >> 1)];
    }
#pragma unroll
    for (int q = 0; q < NT; ++q) {
      const int nc = nW + q * 16 + (l & 15);
#pragma unroll
      for (int j = 0; j < 8; ++j) {
        int kp = kb + ((j < 4) ? (2 * j + 8 * kh) : (16 + 2 * (j - 4) + 8 * kh));
        unsigned int lo = Bs[nc * 66 + kp];
        unsigned int hi = Bs[nc * 66 + kp + 1];
        b.u[j] = lo | (hi << 16);
      }
      acc[q] = __builtin_amdgcn_wmma_f32_16x16x32_bf16(false, a.v, false, b.v,
                                                       (short)0, acc[q], false, false);
    }
  }
  const int row0 = m0 + mT + kh * 8;
#pragma unroll
  for (int q = 0; q < NT; ++q) {
    const int colg = n0 + nW + q * 16 + (l & 15);
    int idx = row0 * ldc + colg;                        // fits in 32-bit here
#pragma unroll
    for (int r = 0; r < 8; ++r) {
      if (OB) ((unsigned short*)C)[idx] = f2bf(acc[q][r]);
      else    ((float*)C)[idx] = acc[q][r];
      idx += ldc;
    }
  }
}

// ------ msg[e,f] += sum_k ee[e,k]*Gn[src,k,f]; last chunk scatters ------
// One thread per (edge, f-pair): all Gn/ee traffic as packed b32 (2 x bf16).
__global__ void __launch_bounds__(256)
k_msg(const unsigned short* __restrict__ EE, const unsigned short* __restrict__ GNC,
      const int* __restrict__ src, const int* __restrict__ dst,
      const float* __restrict__ HBV, float* __restrict__ MSG,
      float* __restrict__ AGG, float* __restrict__ CNT,
      int chunk, int first, int last) {
  int t = blockIdx.x * 256 + threadIdx.x;          // NE*32
  int e = t >> 5, lp = t & 31, f0 = lp * 2;
  int s = src[e];
  const unsigned int* eeu = (const unsigned int*)(EE + (size_t)e * EHID + chunk * KCH);
  const unsigned int* gnu = (const unsigned int*)(GNC + (size_t)s * (KCH * 64)) + lp;
  float a0, a1;
  if (first) {
    const float* hb = HBV + (size_t)s * 64 + f0;
    a0 = hb[0]; a1 = hb[1];
  } else {
    a0 = MSG[e * 64 + f0]; a1 = MSG[e * 64 + f0 + 1];
  }
#pragma unroll
  for (int k2 = 0; k2 < KCH / 2; ++k2) {
    unsigned int ev = eeu[k2];
    float e0 = bf2f((unsigned short)(ev & 0xFFFFu));
    float e1 = bf2f((unsigned short)(ev >> 16));
    unsigned int g0 = gnu[(2 * k2 + 0) * 32];
    unsigned int g1 = gnu[(2 * k2 + 1) * 32];
    a0 += e0 * bf2f((unsigned short)(g0 & 0xFFFFu)) +
          e1 * bf2f((unsigned short)(g1 & 0xFFFFu));
    a1 += e0 * bf2f((unsigned short)(g0 >> 16)) +
          e1 * bf2f((unsigned short)(g1 >> 16));
  }
  if (last) {
    int d = dst[e];
    atomicAdd(&AGG[d * 64 + f0 + 0], a0);
    atomicAdd(&AGG[d * 64 + f0 + 1], a1);
    if (lp == 0) atomicAdd(&CNT[d], 1.0f);
  } else {
    MSG[e * 64 + f0 + 0] = a0;
    MSG[e * 64 + f0 + 1] = a1;
  }
}

// ---- t = relu(h@root + agg/cnt + bias); accumulate BN sum / sumsq ----
__global__ void __launch_bounds__(256)
k_combine(float* __restrict__ T, const float* __restrict__ AGG,
          const float* __restrict__ CNT, const float* __restrict__ bias,
          float* __restrict__ ST) {
  __shared__ float ss[64], sq[64];
  int tid = threadIdx.x;
  if (tid < 64) { ss[tid] = 0.0f; sq[tid] = 0.0f; }
  __syncthreads();
  int t = blockIdx.x * 256 + tid;
  int n = t >> 6, f = t & 63;
  float v = T[t] + AGG[t] / fmaxf(CNT[n], 1.0f) + bias[f];
  v = fmaxf(v, 0.0f);
  T[t] = v;
  atomicAdd(&ss[f], v);
  atomicAdd(&sq[f], v * v);
  __syncthreads();
  if (tid < 64) { atomicAdd(&ST[tid], ss[tid]); atomicAdd(&ST[64 + tid], sq[tid]); }
}

// ---------------- batchnorm apply -> new h (fp32 + bf16) ----------------
__global__ void __launch_bounds__(256)
k_norm(const float* __restrict__ T, const float* __restrict__ ST,
       const float* __restrict__ g, const float* __restrict__ b,
       float* __restrict__ H, unsigned short* __restrict__ HB) {
  int t = blockIdx.x * 256 + threadIdx.x;
  int f = t & 63;
  float mu  = ST[f] * (1.0f / NN);
  float var = ST[64 + f] * (1.0f / NN) - mu * mu;
  float v = (T[t] - mu) * rsqrtf(var + 1e-5f) * g[f] + b[f];
  H[t] = v; HB[t] = f2bf(v);
}

// ---------------- graph pooling (segment sums + counts) ----------------
__global__ void __launch_bounds__(256)
k_pool(const float* __restrict__ H, const int* __restrict__ batch,
       float* __restrict__ P, float* __restrict__ PC) {
  int t = blockIdx.x * 256 + threadIdx.x;
  int n = t >> 6, f = t & 63;
  int g = batch[n];
  atomicAdd(&P[(size_t)g * 64 + f], H[t]);
  if (f == 0) atomicAdd(&PC[g], 1.0f);
}

// ---------------- o = relu((pool[/cnt]) @ LW + LB) ----------------
__global__ void __launch_bounds__(256)
k_head(const float* __restrict__ P, const float* __restrict__ PC, int useMean,
       const float* __restrict__ LW, const float* __restrict__ LB,
       float* __restrict__ O) {
  int t = blockIdx.x * 256 + threadIdx.x;          // NG*64
  int g = t >> 6, f = t & 63;
  float inv = useMean ? (1.0f / fmaxf(PC[g], 1.0f)) : 1.0f;
  float acc = LB[f];
  for (int j = 0; j < 64; ++j)
    acc += P[g * 64 + j] * inv * LW[j * 64 + f];
  O[t] = fmaxf(acc, 0.0f);
}

// ---------------- out = (o1+o2) @ lin2_w + lin2_b ----------------
__global__ void k_out(const float* __restrict__ O1, const float* __restrict__ O2,
                      const float* __restrict__ W, const float* __restrict__ B,
                      float* __restrict__ out) {
  int g = threadIdx.x;
  if (g < NG) {
    float s = 0.0f;
    for (int f = 0; f < 64; ++f) s += (O1[g * 64 + f] + O2[g * 64 + f]) * W[f];
    out[g] = s + B[0];
  }
}

extern "C" void kernel_launch(void* const* d_in, const int* in_sizes, int n_in,
                              void* d_out, int out_size, void* d_ws, size_t ws_size,
                              hipStream_t stream) {
  (void)in_sizes; (void)n_in; (void)out_size; (void)ws_size;
  auto P = [&](int i) { return (const float*)d_in[i]; };
  const float* x  = (const float*)d_in[0];
  const float* ea = (const float*)d_in[1];
  // flatten order: lin0_w=2, lin0_b=3, convs1[l]={w1,b1,w2,b2,root,bias} at 4+6l,
  // convs2 at 22+6l, norm1_{g,b}=40/41, norm2_{g,b}=42/43, lin1=44/45,
  // lin12=46/47, lin2=48/49, edge_index=50, batch=51
  const int* ei    = (const int*)d_in[50];
  const int* batch = (const int*)d_in[51];

  char* wsb = (char*)d_ws;
  size_t off = 0;
  auto alloc = [&](size_t bytes) -> void* {
    void* p = wsb + off;
    off = (off + bytes + 255) & ~(size_t)255;
    return p;
  };
  float*          H    = (float*)alloc((size_t)NN * DIM * 4);
  unsigned short* HB   = (unsigned short*)alloc((size_t)NN * DIM * 2);
  unsigned short* EE   = (unsigned short*)alloc((size_t)NE * EHID * 2);
  unsigned short* W2P  = (unsigned short*)alloc((size_t)64 * 8192 * 2);
  unsigned short* B2P  = (unsigned short*)alloc((size_t)64 * 64 * 2);
  unsigned short* RTP  = (unsigned short*)alloc((size_t)64 * 64 * 2);
  float*          HBV  = (float*)alloc((size_t)NN * DIM * 4);
  float*          ROUT = (float*)alloc((size_t)NN * DIM * 4);
  unsigned short* GNC  = (unsigned short*)alloc((size_t)NN * KCH * 64 * 2);
  float*          MSG  = (float*)alloc((size_t)NE * DIM * 4);
  float*          AGG  = (float*)alloc((size_t)NN * DIM * 4);
  float*          CNT  = (float*)alloc((size_t)NN * 4);
  float*          ST   = (float*)alloc(128 * 4);
  float*          PO   = (float*)alloc((size_t)NG * DIM * 4);
  float*          PC   = (float*)alloc((size_t)NG * 4);
  float*          O1   = (float*)alloc((size_t)NG * DIM * 4);
  float*          O2   = (float*)alloc((size_t)NG * DIM * 4);

  for (int br = 0; br < 2; ++br) {
    k_lin0<<<NN * DIM / 256, 256, 0, stream>>>(x, P(2), P(3), H, HB);
    for (int ly = 0; ly < 3; ++ly) {
      int base = 4 + br * 18 + ly * 6;
      k_edge<<<NE * EHID / 256, 256, 0, stream>>>(ea, P(base), P(base + 1), EE);
      k_pack_w2<<<128 * 4096 / 256, 256, 0, stream>>>(P(base + 2), W2P);
      k_tobf16<<<16, 256, 0, stream>>>(P(base + 3), B2P, 4096);
      k_tobf16<<<16, 256, 0, stream>>>(P(base + 4), RTP, 4096);
      // hb = h @ reshape(b2)   and   rootout = h @ root  (fp32 outputs)
      k_gemm64<1, false><<<dim3(1, NN / 32), 256, 0, stream>>>(HB, B2P, 0, 64, HBV, 64);
      k_gemm64<1, false><<<dim3(1, NN / 32), 256, 0, stream>>>(HB, RTP, 0, 64, ROUT, 64);
      k_zero<<<NN * DIM / 256, 256, 0, stream>>>(AGG, NN * DIM);
      k_zero<<<NN / 256, 256, 0, stream>>>(CNT, NN);
      for (int c = 0; c < NCH; ++c) {
        // Gn chunk: [8192, 2048] bf16 (33.5 MB, L2-resident)
        k_gemm64<4, true><<<dim3(KCH * 64 / 256, NN / 32), 256, 0, stream>>>(
            HB, W2P, c * KCH * 64, 8192, GNC, KCH * 64);
        k_msg<<<NE * 32 / 256, 256, 0, stream>>>(
            EE, GNC, ei, ei + NE, HBV, MSG, AGG, CNT, c, c == 0, c == NCH - 1);
      }
      k_zero<<<1, 256, 0, stream>>>(ST, 128);
      k_combine<<<NN * DIM / 256, 256, 0, stream>>>(ROUT, AGG, CNT, P(base + 5), ST);
      k_norm<<<NN * DIM / 256, 256, 0, stream>>>(ROUT, ST, P(40 + br * 2),
                                                 P(41 + br * 2), H, HB);
    }
    k_zero<<<NG * DIM / 256, 256, 0, stream>>>(PO, NG * DIM);
    k_zero<<<1, 64, 0, stream>>>(PC, NG);
    k_pool<<<NN * DIM / 256, 256, 0, stream>>>(H, batch, PO, PC);
    k_head<<<NG * DIM / 256, 256, 0, stream>>>(PO, PC, br, P(44 + br * 2),
                                               P(45 + br * 2), br ? O2 : O1);
  }
  k_out<<<1, 64, 0, stream>>>(O1, O2, P(48), P(49), (float*)d_out);
}
